// torch_RNN_full_manual_6940667150563
// MI455X (gfx1250) — matrix operand order repaired and checked by reading, not compiled, not run on hardware
//
#include <hip/hip_runtime.h>
#include <hip/hip_bf16.h>

// ---------------------------------------------------------------------------
// Vanilla RNN  h_t = relu(h_{t-1} @ Wh^T + x_t @ Wx^T),  out = h_T @ Wo^T
// B=256 T=512 IN=64 H=1024 OUT=128
//
// Strategy (MI455X / gfx1250, wave32, WMMA):
//  * One persistent kernel runs all 512 recurrent steps. 4 independent batch
//    slices (clusters) x 16 workgroups; per-step sync = CDNA5 cluster barrier
//    (s_barrier_signal/-wait -3 via __builtin_amdgcn_s_cluster_barrier) plus
//    an L2-atomic arrival counter as the launch-config-independent fallback.
//  * Weights pre-packed to f16 WMMA B-fragment layout; each WG DMAs its
//    64-column Wh slice (128KB) + Wx slice (8KB) into LDS once via the
//    Tensor Data Mover (tensor_load_to_lds, TENSORcnt), reused 512x.
//  * x-projection fused into the step GEMM as 2 extra K-tiles (K=1088).
//  * Math: v_wmma_f32_16x16x32_f16 (f16 inputs, f32 accumulate).
// ---------------------------------------------------------------------------

#define B_    256
#define T_    512
#define IN_   64
#define H_    1024
#define OUT_  128

#define NCLUST        4
#define WG_PER_CLUST  16
#define BC            (B_ / NCLUST)   // 64 batch rows per cluster
#define THREADS       128             // 4 waves

typedef _Float16 half_t;
typedef __attribute__((ext_vector_type(16))) _Float16 v16h;
typedef __attribute__((ext_vector_type(8)))  _Float16 v8h;
typedef __attribute__((ext_vector_type(8)))  float    v8f;
typedef __attribute__((ext_vector_type(4)))  unsigned int u32x4;
typedef __attribute__((ext_vector_type(8)))  int i32x8;
typedef __attribute__((ext_vector_type(4)))  int i32x4;

// ---- workspace layout (bytes) ----
#define OFF_CTR  0u
#define OFF_WH   256u                                   // 1024*1024 f16 = 2MB
#define OFF_WX   (OFF_WH + 2u*1024u*1024u)              // 1024*64  f16 = 128KB
#define OFF_WO   (OFF_WX + 128u*1024u)                  // 128*1024 f16 = 256KB
#define OFF_HA   (OFF_WO + 256u*1024u)                  // 256*1024 f16 = 512KB
#define OFF_HB   (OFF_HA + 512u*1024u)
// total ~3.4MB

#define SMEM_HALVES (65536 + 4096)                      // Wh slice + Wx slice
#define SMEM_BYTES  (SMEM_HALVES * 2)

// Toolchain hedge: the therock-10.0 headers (6-arg tensor_load_to_lds) ship
// amd_gfx1250_TDM.h; ROCm 7.2 (5-arg builtin) does not.
#if defined(__has_include)
#  if __has_include(<hip/amd_detail/amd_gfx1250_TDM.h>)
#    define TDM_6ARG 1
#  endif
#endif

// ---------------------------------------------------------------------------
// helpers
// ---------------------------------------------------------------------------

__device__ __forceinline__ v8f wmma16(v16h a, v16h b, v8f c) {
  // (neg_a, A, neg_b, B, c_mod, C, reuse_a, reuse_b)
  return __builtin_amdgcn_wmma_f32_16x16x32_f16(false, a, false, b,
                                                (short)0, c, false, false);
}

// A fragment (16x32 f16): lane L (row M = L&15) holds halves 0..7 = K[kb..kb+7]
// and halves 8..15 = K[kb+16..kb+23], kb = k0 + (L>=16 ? 8 : 0).
// p points at (row, k0+koff); chunks at p and p+16.
__device__ __forceinline__ v16h load_a16(const half_t* p) {
  v8h lo = *(const v8h*)(p);
  v8h hi = *(const v8h*)(p + 16);
  return __builtin_shufflevector(lo, hi, 0,1,2,3,4,5,6,7,8,9,10,11,12,13,14,15);
}

// Same A layout but sourced from fp32 (for the fused x_t contribution).
__device__ __forceinline__ v16h load_a32(const float* p) {
  v16h a;
#pragma unroll
  for (int i = 0; i < 8; ++i) {
    a[i]     = (half_t)p[i];
    a[8 + i] = (half_t)p[16 + i];
  }
  return a;
}

// B fragment: pre-packed contiguous 16 halves per lane.
__device__ __forceinline__ v16h load_b16(const half_t* p) {
  v8h lo = *(const v8h*)(p);
  v8h hi = *(const v8h*)(p + 8);
  return __builtin_shufflevector(lo, hi, 0,1,2,3,4,5,6,7,8,9,10,11,12,13,14,15);
}

// C/D layout: VGPR r, lanes 0-15 -> (M=r, N=lane); lanes 16-31 -> (M=8+r, N=lane-16)
__device__ __forceinline__ void store_relu_f16(v8f c, half_t* h,
                                               int m0, int n0, int lane) {
  const int col = n0 + (lane & 15);
  const int rb  = m0 + ((lane >> 4) * 8);
#pragma unroll
  for (int r = 0; r < 8; ++r) {
    float v = c[r];
    v = v > 0.f ? v : 0.f;
    h[(size_t)(rb + r) * H_ + col] = (half_t)v;
  }
}

__device__ __forceinline__ void store_f32(v8f c, float* o,
                                          int m0, int n0, int lane) {
  const int col = n0 + (lane & 15);
  const int rb  = m0 + ((lane >> 4) * 8);
#pragma unroll
  for (int r = 0; r < 8; ++r) o[(size_t)(rb + r) * OUT_ + col] = c[r];
}

// TDM: DMA a contiguous `bytes` block (8-byte elements, 1-D tile) from global
// memory into this workgroup's LDS at byte offset `lds_off`.
// D# group0: [1:0]=count=1, lds_addr, global_addr[56:0], type=2 ("image").
// D# group1: data_size=3 (8B), tensor_dim0=tile_dim0=bytes/8, tensor_dim1=1,
//            tile_dim1=1, tensor_dim0_stride=bytes/8, no pad/iterate/multicast.
__device__ __forceinline__ void tdm_load_1d(const void* src, unsigned lds_off,
                                            unsigned bytes) {
  const unsigned units = bytes >> 3;                 // 8-byte data_size units
  const unsigned long long ga = (unsigned long long)src;
  u32x4 g0 = { 1u,                                   // count=1, user descriptor
               lds_off,                              // lds_addr (bytes)
               (unsigned)(ga & 0xFFFFFFFFu),         // global_addr[31:0]
               (unsigned)((ga >> 32) & 0x1FFFFFFu) | (2u << 30) }; // [56:32]|type=2
  i32x8 g1 = { (int)(3u << 16),                      // data_size=3 (8 bytes)
               (int)((units & 0xFFFFu) << 16),       // tensor_dim0[15:0]
               (int)(((units >> 16) & 0xFFFFu) | (1u << 16)), // dim0[31:16], dim1=1
               (int)((units & 0xFFFFu) << 16),       // tile_dim0 (<=65535)
               1,                                    // tile_dim1=1
               (int)units,                           // tensor_dim0_stride[31:0]
               0, 0 };
  i32x4 g2 = { 0, 0, 0, 0 };
  i32x4 g3 = { 0, 0, 0, 0 };
#ifdef TDM_6ARG
  i32x8 g4 = { 0, 0, 0, 0, 0, 0, 0, 0 };
  __builtin_amdgcn_tensor_load_to_lds(g0, g1, g2, g3, g4, 0);
#else
  __builtin_amdgcn_tensor_load_to_lds(g0, g1, g2, g3, 0);
#endif
}

// Per-step barrier across the 16 WGs of a cluster.
// Hardware path: CDNA5 cluster barrier (NOP if dispatch not clustered).
// Portable path: monotonically increasing per-cluster arrival counter in L2.
__device__ __forceinline__ void step_barrier(unsigned* ctr, unsigned target) {
  __syncthreads();
  if (threadIdx.x == 0) {   // one arrival per workgroup, release prior stores
    __hip_atomic_fetch_add(ctr, 1u, __ATOMIC_RELEASE, __HIP_MEMORY_SCOPE_AGENT);
  }
  if ((threadIdx.x >> 5) == 0) {     // one wave per WG signals+waits (-3)
    __builtin_amdgcn_s_cluster_barrier();
  }
  if (threadIdx.x == 0) {
    while (__hip_atomic_load(ctr, __ATOMIC_ACQUIRE,
                             __HIP_MEMORY_SCOPE_AGENT) < target)
      __builtin_amdgcn_s_sleep(1);
  }
  __syncthreads();
}

// ---------------------------------------------------------------------------
// weight packing: fragment order so the hot loop's B load is one contiguous
// 32B LDS/L2 read per lane.
//   index = (((jt*4 + nt)*KS + ks)*512 + lane*16 + h)
//   element = W[jt*64 + nt*16 + (lane&15)][ks*32 + (lane>=16?8:0) + (h&7) + (h>=8?16:0)]
// ---------------------------------------------------------------------------

__global__ void pack_wh_kernel(const float* __restrict__ Wh,
                               half_t* __restrict__ whp) {
  unsigned idx = blockIdx.x * 256u + threadIdx.x;      // 2^20 total
  int h  = idx & 15;
  int l  = (idx >> 4)  & 31;
  int ks = (idx >> 9)  & 31;
  int nt = (idx >> 14) & 3;
  int jt = idx >> 16;                                  // 0..15
  int j  = jt * 64 + nt * 16 + (l & 15);
  int k  = ks * 32 + ((l >> 4) * 8) + (h & 7) + ((h >> 3) * 16);
  whp[idx] = (half_t)Wh[(size_t)j * H_ + k];
}

__global__ void pack_wx_kernel(const float* __restrict__ Wx,
                               half_t* __restrict__ wxp,
                               unsigned* __restrict__ ctr) {
  if (blockIdx.x == 0 && threadIdx.x < 8) ctr[threadIdx.x] = 0u;  // reset barrier
  unsigned idx = blockIdx.x * 256u + threadIdx.x;      // 2^16 total
  int h  = idx & 15;
  int l  = (idx >> 4)  & 31;
  int ks = (idx >> 9)  & 1;
  int nt = (idx >> 10) & 3;
  int jt = idx >> 12;                                  // 0..15
  int j  = jt * 64 + nt * 16 + (l & 15);
  int k  = ks * 32 + ((l >> 4) * 8) + (h & 7) + ((h >> 3) * 16);
  wxp[idx] = (half_t)Wx[(size_t)j * IN_ + k];
}

__global__ void pack_wo_kernel(const float* __restrict__ Wo,
                               half_t* __restrict__ wop) {
  unsigned idx = blockIdx.x * 256u + threadIdx.x;      // 2^17 total
  int h  = idx & 15;
  int l  = (idx >> 4)  & 31;
  int ks = (idx >> 9)  & 31;
  int nt = (idx >> 14) & 3;
  int jt = idx >> 16;                                  // 0..1
  int j  = jt * 64 + nt * 16 + (l & 15);
  int k  = ks * 32 + ((l >> 4) * 8) + (h & 7) + ((h >> 3) * 16);
  wop[idx] = (half_t)Wo[(size_t)j * H_ + k];
}

// ---------------------------------------------------------------------------
// persistent recurrence kernel: grid = 64 WGs (4 clusters x 16), block = 128
// ---------------------------------------------------------------------------

__global__ void rnn_persistent_kernel(const float* __restrict__ x,
                                      const half_t* __restrict__ whp,
                                      const half_t* __restrict__ wxp,
                                      half_t* __restrict__ hA,
                                      half_t* __restrict__ hB,
                                      unsigned* __restrict__ ctrs) {
  const int lane  = threadIdx.x & 31;
  const int wave  = threadIdx.x >> 5;
  const int wg    = blockIdx.x % WG_PER_CLUST;   // hidden-column tile (64 cols)
  const int cl    = blockIdx.x / WG_PER_CLUST;   // batch slice
  const int bbase = cl * BC;
  const int jbase = wg * 64;
  unsigned* ctr   = ctrs + cl;

  extern __shared__ __align__(16) half_t smem[];
  half_t* s_wh = smem;            // 65536 halves: (nt*32+ks)*512 + lane*16 + h
  half_t* s_wx = smem + 65536;    //  4096 halves: (nt*2 +ks)*512 + lane*16 + h

  // --- one-time: TDM-DMA this WG's packed weight slices into LDS ---
  if (wave == 0) {
    tdm_load_1d(whp + (size_t)wg * 65536, 0u,      131072u);  // s_wh
    tdm_load_1d(wxp + (size_t)wg * 4096,  131072u, 8192u);    // s_wx
  }

  // --- one-time (overlaps TDM): zero h_{-1}, 4 full rows of hA per WG ---
  {
    const float4 z = {0.f, 0.f, 0.f, 0.f};
    for (int r = 0; r < 4; ++r) {
      int row = bbase + wg * 4 + r;
      float4* p = (float4*)(hA + (size_t)row * H_);
      for (int c = threadIdx.x; c < H_ / 8; c += THREADS) p[c] = z;
    }
  }

  if (wave == 0) __builtin_amdgcn_s_wait_tensorcnt(0);  // LDS slices resident

  unsigned phase = 0;
  step_barrier(ctr, ++phase * WG_PER_CLUST);   // init done across cluster

  // wave -> 2x2 grid of 16x16 output fragments
  const int msel = (wave & 1) * 2;             // M strips {msel, msel+1}
  const int nsel = (wave >> 1) * 2;            // N tiles  {nsel, nsel+1}
  const int rA0  = bbase + (msel + 0) * 16 + (lane & 15);
  const int rA1  = bbase + (msel + 1) * 16 + (lane & 15);
  const int koff = (lane >> 4) * 8;            // A per-lane K offset
  const int bl   = lane * 16;                  // B per-lane LDS offset (halves)

  for (int t = 0; t < T_; ++t) {
    const half_t* hcur = (t & 1) ? hB : hA;
    half_t*       hnxt = (t & 1) ? hA : hB;

    v8f c00 = {}, c01 = {}, c10 = {}, c11 = {};

    // K over H (recurrent term)
    for (int ks = 0; ks < 32; ++ks) {
      const int k0 = ks * 32 + koff;
      v16h a0 = load_a16(hcur + (size_t)rA0 * H_ + k0);
      v16h a1 = load_a16(hcur + (size_t)rA1 * H_ + k0);
      v16h b0 = load_b16(s_wh + ((nsel + 0) * 32 + ks) * 512 + bl);
      v16h b1 = load_b16(s_wh + ((nsel + 1) * 32 + ks) * 512 + bl);
      c00 = wmma16(a0, b0, c00);
      c01 = wmma16(a0, b1, c01);
      c10 = wmma16(a1, b0, c10);
      c11 = wmma16(a1, b1, c11);
    }

    // K over IN (fused x-projection), f32 -> f16 on the fly
    const float* xr0 = x + ((size_t)rA0 * T_ + t) * IN_;
    const float* xr1 = x + ((size_t)rA1 * T_ + t) * IN_;
#pragma unroll
    for (int ks = 0; ks < 2; ++ks) {
      const int k0 = ks * 32 + koff;
      v16h a0 = load_a32(xr0 + k0);
      v16h a1 = load_a32(xr1 + k0);
      v16h b0 = load_b16(s_wx + ((nsel + 0) * 2 + ks) * 512 + bl);
      v16h b1 = load_b16(s_wx + ((nsel + 1) * 2 + ks) * 512 + bl);
      c00 = wmma16(a0, b0, c00);
      c01 = wmma16(a0, b1, c01);
      c10 = wmma16(a1, b0, c10);
      c11 = wmma16(a1, b1, c11);
    }

    if (t + 1 < T_) {                 // warm L2/WGP$ for next timestep's x
      __builtin_prefetch(xr0 + IN_, 0, 1);
      __builtin_prefetch(xr1 + IN_, 0, 1);
    }

    // relu + f16 writeback of this WG's 64x64 tile of h_{t}
    store_relu_f16(c00, hnxt, bbase + (msel + 0) * 16, jbase + (nsel + 0) * 16, lane);
    store_relu_f16(c01, hnxt, bbase + (msel + 0) * 16, jbase + (nsel + 1) * 16, lane);
    store_relu_f16(c10, hnxt, bbase + (msel + 1) * 16, jbase + (nsel + 0) * 16, lane);
    store_relu_f16(c11, hnxt, bbase + (msel + 1) * 16, jbase + (nsel + 1) * 16, lane);

    step_barrier(ctr, ++phase * WG_PER_CLUST);
  }
  // T=512 even -> final h lives in hA
}

// ---------------------------------------------------------------------------
// output GEMM: out[256,128] = h_final(f16) @ Wo^T, f32 out.
// grid = 8 WGs (4 M-tiles x 2 N-tiles of 64), block = 128.
// ---------------------------------------------------------------------------

__global__ void rnn_out_kernel(const half_t* __restrict__ hfin,
                               const half_t* __restrict__ wop,
                               float* __restrict__ out) {
  const int lane = threadIdx.x & 31;
  const int wave = threadIdx.x >> 5;
  const int mt   = blockIdx.x >> 1;          // 0..3
  const int ntt  = blockIdx.x & 1;           // 0..1
  const int msel = (wave & 1) * 2;
  const int nsel = (wave >> 1) * 2;
  const int r0   = mt * 64 + (msel + 0) * 16 + (lane & 15);
  const int r1   = r0 + 16;
  const int koff = (lane >> 4) * 8;
  const half_t* bp = wop + (size_t)ntt * 65536;

  v8f c00 = {}, c01 = {}, c10 = {}, c11 = {};
  for (int ks = 0; ks < 32; ++ks) {
    const int k0 = ks * 32 + koff;
    v16h a0 = load_a16(hfin + (size_t)r0 * H_ + k0);
    v16h a1 = load_a16(hfin + (size_t)r1 * H_ + k0);
    v16h b0 = load_b16(bp + (size_t)((nsel + 0) * 32 + ks) * 512 + lane * 16);
    v16h b1 = load_b16(bp + (size_t)((nsel + 1) * 32 + ks) * 512 + lane * 16);
    c00 = wmma16(a0, b0, c00);
    c01 = wmma16(a0, b1, c01);
    c10 = wmma16(a1, b0, c10);
    c11 = wmma16(a1, b1, c11);
  }
  store_f32(c00, out, mt * 64 + (msel + 0) * 16, ntt * 64 + (nsel + 0) * 16, lane);
  store_f32(c01, out, mt * 64 + (msel + 0) * 16, ntt * 64 + (nsel + 1) * 16, lane);
  store_f32(c10, out, mt * 64 + (msel + 1) * 16, ntt * 64 + (nsel + 0) * 16, lane);
  store_f32(c11, out, mt * 64 + (msel + 1) * 16, ntt * 64 + (nsel + 1) * 16, lane);
}

// ---------------------------------------------------------------------------

extern "C" void kernel_launch(void* const* d_in, const int* in_sizes, int n_in,
                              void* d_out, int out_size, void* d_ws, size_t ws_size,
                              hipStream_t stream) {
  (void)in_sizes; (void)n_in; (void)out_size; (void)ws_size;

  const float* x  = (const float*)d_in[0];   // [B,T,IN]
  const float* Wh = (const float*)d_in[1];   // [H,H]
  const float* Wx = (const float*)d_in[2];   // [H,IN]
  const float* Wo = (const float*)d_in[3];   // [OUT,H]
  float* out = (float*)d_out;                // [B,OUT]

  char* ws = (char*)d_ws;
  unsigned* ctr = (unsigned*)(ws + OFF_CTR);
  half_t* whp = (half_t*)(ws + OFF_WH);
  half_t* wxp = (half_t*)(ws + OFF_WX);
  half_t* wop = (half_t*)(ws + OFF_WO);
  half_t* hA  = (half_t*)(ws + OFF_HA);
  half_t* hB  = (half_t*)(ws + OFF_HB);

  pack_wh_kernel<<<4096, 256, 0, stream>>>(Wh, whp);
  pack_wx_kernel<<<256,  256, 0, stream>>>(Wx, wxp, ctr);  // also resets barrier ctrs
  pack_wo_kernel<<<512,  256, 0, stream>>>(Wo, wop);

  rnn_persistent_kernel<<<NCLUST * WG_PER_CLUST, THREADS, SMEM_BYTES, stream>>>(
      x, whp, wxp, hA, hB, ctr);

  rnn_out_kernel<<<8, THREADS, 0, stream>>>(hA, wop, out);
}
